// SNN3_21878563406062
// MI455X (gfx1250) — matrix operand order repaired and checked by reading, not compiled, not run on hardware
//
#include <hip/hip_runtime.h>
#include <stdint.h>

// ---------------------------------------------------------------------------
// SNN (3-layer, T=50 scan) for MI455X / gfx1250, wave32 + WMMA bf16.
//
// ~59.4 GFLOP vs ~508 MB streamed input (117 FLOP/B; 23.3 TB/s floor ~22us).
// GEMMs on v_wmma_f32_16x16x32_bf16 (f32 accumulate). Weights packed once
// into bf16 B-fragment layout in d_ws (L2-resident, ~317 KB).
//
// Per block: one 16-row batch tile, 4 waves, loop over all 50 timesteps.
//  - x_t in LDS as fp32; x_{t+1} prefetched into the SAME buffer with
//    GLOBAL_LOAD_ASYNC_TO_LDS_B128 (ASYNCcnt), overlapping step-t compute.
//  - L1 K-loop split in two halves (10 A-frags live at a time, 4 persistent
//    accumulators) to keep VGPR allocation < 256 so the compiler can
//    multi-buffer the streaming B loads (no s_set_vgpr_msb churn, no
//    wait-all before each wmma).
//  - Membrane state m1/m2 in per-wave registers (tile ownership static
//    across t); only spikes s1/s2 (next layer's A operand) are in LDS.
// ---------------------------------------------------------------------------

typedef __attribute__((ext_vector_type(16))) __bf16 v16bf;
typedef __attribute__((ext_vector_type(8)))  __bf16 v8bf;
typedef __attribute__((ext_vector_type(8)))  float  v8f;

#define T_STEPS 50
#define IN_DIM  620
#define H1      200
#define H2      100
#define OUT_DIM 11

#define K1P 640
#define K2P 224
#define K3P 128

#define NK1 (K1P/32)  // 20
#define NT1 13
#define NK2 (K2P/32)  // 7
#define NT2 7
#define NK3 (K3P/32)  // 4
#define NKH 10        // half of NK1

// ---------------------------------------------------------------------------
// Weight fragment packer: W [outDim, inDim] fp32 -> bf16 B-matrix fragments.
// Fragment f = nt*nK + ks: lane l<16 -> column n=nt*16+l, K=ks*32+{0..7,16..23};
// lane l>=16 -> column n=nt*16+l-16, K=ks*32+{8..15,24..31}. 32 B contiguous per lane.
// ---------------------------------------------------------------------------
__global__ void snn_pack_weights(const float* __restrict__ W,
                                 __bf16* __restrict__ dst,
                                 int outDim, int inDim, int nK, int nFrag) {
    int id = blockIdx.x * blockDim.x + threadIdx.x;
    int total = nFrag * 512;
    if (id >= total) return;
    int f    = id >> 9;
    int r    = id & 511;
    int lane = r >> 4;
    int j    = r & 15;
    int nt = f / nK;
    int ks = f - nt * nK;
    int n = nt * 16 + (lane & 15);
    int k = ks * 32 + ((lane >> 4) << 3) + (j & 7) + ((j >> 3) << 4);
    float v = (n < outDim && k < inDim) ? W[(size_t)n * inDim + k] : 0.0f;
    dst[(size_t)f * 512 + r] = (__bf16)v;
}

__device__ __forceinline__ float swishf(float x) {
    return x / (1.0f + __expf(-10.0f * x));   // x * sigmoid(10x)
}

__device__ __forceinline__ void wait_async0() {
#if __has_builtin(__builtin_amdgcn_s_wait_asynccnt)
    __builtin_amdgcn_s_wait_asynccnt(0);
#else
    asm volatile("s_wait_asynccnt 0x0" ::: "memory");
#endif
}

__device__ __forceinline__ void wait_ds0() {
#if __has_builtin(__builtin_amdgcn_s_wait_dscnt)
    __builtin_amdgcn_s_wait_dscnt(0);
#else
    asm volatile("s_wait_dscnt 0x0" ::: "memory");
#endif
}

// Async prefetch of one x_t tile (16 rows x 620 fp32) into LDS (row stride
// 640 fp32; pad cols pre-zeroed). 8 threads per row, pure pointer-increment
// address stream; per lane 19-20 GLOBAL_LOAD_ASYNC_TO_LDS_B128 (ASYNCcnt).
__device__ __forceinline__ void stage_async(const float* __restrict__ x,
                                            float* __restrict__ xf,
                                            int b0, int t, int tid) {
    const int l   = tid & 7;
    const int row = tid >> 3;
    uint64_t ga = (uint64_t)(x + ((size_t)(b0 + row) * T_STEPS + t) * IN_DIM + l * 4);
    uint32_t la = (uint32_t)(size_t)xf + (uint32_t)((row * K1P + l * 4) * 4);
#pragma unroll 1
    for (int j = 0; j < 19; ++j) {        // cols l*4 + 32*j, full passes
        asm volatile("global_load_async_to_lds_b128 %0, %1, off"
                     :: "v"(la), "v"(ga) : "memory");
        ga += 128;
        la += 128;
    }
    if (l < 3) {                          // tail: 155 = 8*19 + 3 float4/row
        asm volatile("global_load_async_to_lds_b128 %0, %1, off"
                     :: "v"(la), "v"(ga) : "memory");
    }
}

// Load one 16x32 bf16 A-fragment from an fp32 LDS tile (convert on the fly).
__device__ __forceinline__ v16bf a_frag_f32(const float* base) {
    const float4 f0 = *(const float4*)(base);
    const float4 f1 = *(const float4*)(base + 4);
    const float4 f2 = *(const float4*)(base + 16);
    const float4 f3 = *(const float4*)(base + 20);
    v16bf a;
    a[0]=(__bf16)f0.x;  a[1]=(__bf16)f0.y;  a[2]=(__bf16)f0.z;  a[3]=(__bf16)f0.w;
    a[4]=(__bf16)f1.x;  a[5]=(__bf16)f1.y;  a[6]=(__bf16)f1.z;  a[7]=(__bf16)f1.w;
    a[8]=(__bf16)f2.x;  a[9]=(__bf16)f2.y;  a[10]=(__bf16)f2.z; a[11]=(__bf16)f2.w;
    a[12]=(__bf16)f3.x; a[13]=(__bf16)f3.y; a[14]=(__bf16)f3.z; a[15]=(__bf16)f3.w;
    return a;
}

// Load one 16x32 bf16 A-fragment from a bf16 LDS tile.
__device__ __forceinline__ v16bf a_frag_bf16(const __bf16* ap) {
    v8bf alo = *(const v8bf*)ap;
    v8bf ahi = *(const v8bf*)(ap + 16);
    v16bf a;
#pragma unroll
    for (int q = 0; q < 8; ++q) { a[q] = alo[q]; a[q + 8] = ahi[q]; }
    return a;
}

// fc = swish(gemm); m = m*0.2*(1-s_prev) + fc; s = swish(m).
__device__ __forceinline__ void mem_update_tile(const v8f& acc, v8f& mreg,
                                                __bf16* sLds, int stride,
                                                int col, int rowhi) {
#pragma unroll
    for (int i = 0; i < 8; ++i) {
        int rowc = i + rowhi;
        float fc = swishf(acc[i]);
        float sp = (float)sLds[rowc * stride + col];
        float m  = mreg[i] * 0.2f * (1.0f - sp) + fc;
        mreg[i] = m;
        sLds[rowc * stride + col] = (__bf16)swishf(m);
    }
}

__global__ void __launch_bounds__(128)
snn_forward(const float* __restrict__ x,
            const __bf16* __restrict__ wf1,
            const __bf16* __restrict__ wf2,
            const __bf16* __restrict__ wf3,
            float* __restrict__ out) {
    __shared__ float  xf[16 * K1P];   // x_t tile, fp32, zero-padded K (40 KB)
    __shared__ __bf16 s1[16 * K2P];   // layer-1 spikes (layer-2 A input)
    __shared__ __bf16 s2[16 * K3P];   // layer-2 spikes (layer-3 A input)

    const int tid  = threadIdx.x;
    const int wave = tid >> 5;
    const int lane = tid & 31;
    const int b0   = blockIdx.x * 16;

    // Zero xf K-padding (staging never writes cols 620..639) and spike state.
    for (int i = tid; i < 16 * 20; i += 128) {
        int r = i / 20, c = 620 + (i - r * 20);
        xf[r * K1P + c] = 0.0f;
    }
    for (int i = tid; i < 16 * K2P; i += 128) s1[i] = (__bf16)0.0f;
    for (int i = tid; i < 16 * K3P; i += 128) s2[i] = (__bf16)0.0f;

    stage_async(x, xf, b0, 0, tid);       // prefetch x_0 (disjoint from zeros)
    wait_async0();
    __syncthreads();

    // Per-wave register membrane state (tile ownership static across t).
    // L1 tiles: wave, wave+4, wave+8 (all valid), wave+12 (wave 0 only).
    v8f m1r0 = {}, m1r1 = {}, m1r2 = {}, m1r3 = {};
    // L2 tiles: wave (valid), wave+4 (waves 0..2).
    v8f m2r0 = {}, m2r1 = {};
    v8f m3 = {}, sum3 = {};

    const int rowA  = lane & 15;
    const int aoff  = (lane >> 4) * 8;
    const int rowhi = (lane >> 4) << 3;
    const int colw  = lane & 15;

    const __bf16* wl0 = wf1 + (size_t)(wave)      * NK1 * 512 + lane * 16;
    const __bf16* wl1 = wf1 + (size_t)(wave + 4)  * NK1 * 512 + lane * 16;
    const __bf16* wl2 = wf1 + (size_t)(wave + 8)  * NK1 * 512 + lane * 16;
    const __bf16* wl3 = wf1 + (size_t)12          * NK1 * 512 + lane * 16;
    const __bf16* wm0 = wf2 + (size_t)(wave)      * NK2 * 512 + lane * 16;
    const __bf16* wm1 = wf2 + (size_t)(wave + 4)  * NK2 * 512 + lane * 16;
    const __bf16* wn0 = wf3 + lane * 16;

    for (int t = 0; t < T_STEPS; ++t) {
        v8f acc0 = {}, acc1 = {}, acc2 = {}, acc3 = {};

        // ---- layer 1, first K half: frags 0..9
        {
            v16bf a1[NKH];
#pragma unroll
            for (int k = 0; k < NKH; ++k)
                a1[k] = a_frag_f32(&xf[rowA * K1P + k * 32 + aoff]);
#pragma unroll
            for (int k = 0; k < NKH; ++k) {
                v16bf b0v = *(const v16bf*)(wl0 + (size_t)k * 512);
                acc0 = __builtin_amdgcn_wmma_f32_16x16x32_bf16(false, a1[k], false, b0v, (short)0, acc0, false, false);
                v16bf b1v = *(const v16bf*)(wl1 + (size_t)k * 512);
                acc1 = __builtin_amdgcn_wmma_f32_16x16x32_bf16(false, a1[k], false, b1v, (short)0, acc1, false, false);
                v16bf b2v = *(const v16bf*)(wl2 + (size_t)k * 512);
                acc2 = __builtin_amdgcn_wmma_f32_16x16x32_bf16(false, a1[k], false, b2v, (short)0, acc2, false, false);
            }
            if (wave == 0) {
#pragma unroll
                for (int k = 0; k < NKH; ++k) {
                    v16bf b3v = *(const v16bf*)(wl3 + (size_t)k * 512);
                    acc3 = __builtin_amdgcn_wmma_f32_16x16x32_bf16(false, a1[k], false, b3v, (short)0, acc3, false, false);
                }
            }
        }

        // ---- layer 1, second K half: frags 10..19; then release xf
        {
            v16bf a1[NKH];
#pragma unroll
            for (int k = 0; k < NKH; ++k)
                a1[k] = a_frag_f32(&xf[rowA * K1P + (NKH + k) * 32 + aoff]);
            wait_ds0();               // A data in registers on every wave...
            __syncthreads();          // ...before anyone overwrites xf
            if (t + 1 < T_STEPS)      // prefetch x_{t+1}: overlaps L1b/L2/L3
                stage_async(x, xf, b0, t + 1, tid);
#pragma unroll
            for (int k = 0; k < NKH; ++k) {
                v16bf b0v = *(const v16bf*)(wl0 + (size_t)(NKH + k) * 512);
                acc0 = __builtin_amdgcn_wmma_f32_16x16x32_bf16(false, a1[k], false, b0v, (short)0, acc0, false, false);
                v16bf b1v = *(const v16bf*)(wl1 + (size_t)(NKH + k) * 512);
                acc1 = __builtin_amdgcn_wmma_f32_16x16x32_bf16(false, a1[k], false, b1v, (short)0, acc1, false, false);
                v16bf b2v = *(const v16bf*)(wl2 + (size_t)(NKH + k) * 512);
                acc2 = __builtin_amdgcn_wmma_f32_16x16x32_bf16(false, a1[k], false, b2v, (short)0, acc2, false, false);
            }
            if (wave == 0) {
#pragma unroll
                for (int k = 0; k < NKH; ++k) {
                    v16bf b3v = *(const v16bf*)(wl3 + (size_t)(NKH + k) * 512);
                    acc3 = __builtin_amdgcn_wmma_f32_16x16x32_bf16(false, a1[k], false, b3v, (short)0, acc3, false, false);
                }
            }
        }

        // ---- layer 1 membrane/spike update
        mem_update_tile(acc0, m1r0, s1, K2P, (wave)      * 16 + colw, rowhi);
        mem_update_tile(acc1, m1r1, s1, K2P, (wave + 4)  * 16 + colw, rowhi);
        mem_update_tile(acc2, m1r2, s1, K2P, (wave + 8)  * 16 + colw, rowhi);
        if (wave == 0)
            mem_update_tile(acc3, m1r3, s1, K2P, 12 * 16 + colw, rowhi);
        __syncthreads();

        // ---- layer 2: [16,224] x [224,112]
        {
            v16bf a2[NK2];
#pragma unroll
            for (int k = 0; k < NK2; ++k)
                a2[k] = a_frag_bf16(&s1[rowA * K2P + k * 32 + aoff]);
            v8f c0 = {}, c1 = {};
#pragma unroll
            for (int k = 0; k < NK2; ++k) {
                v16bf b0v = *(const v16bf*)(wm0 + (size_t)k * 512);
                c0 = __builtin_amdgcn_wmma_f32_16x16x32_bf16(false, a2[k], false, b0v, (short)0, c0, false, false);
            }
            if (wave < 3) {
#pragma unroll
                for (int k = 0; k < NK2; ++k) {
                    v16bf b1v = *(const v16bf*)(wm1 + (size_t)k * 512);
                    c1 = __builtin_amdgcn_wmma_f32_16x16x32_bf16(false, a2[k], false, b1v, (short)0, c1, false, false);
                }
            }
            mem_update_tile(c0, m2r0, s2, K3P, (wave) * 16 + colw, rowhi);
            if (wave < 3)
                mem_update_tile(c1, m2r1, s2, K3P, (wave + 4) * 16 + colw, rowhi);
        }
        __syncthreads();

        // ---- layer 3: [16,128] x [128,16], wave 0; state in VGPRs
        if (wave == 0) {
            v8f acc = {};
#pragma unroll
            for (int k = 0; k < NK3; ++k) {
                v16bf a = a_frag_bf16(&s2[rowA * K3P + k * 32 + aoff]);
                v16bf b = *(const v16bf*)(wn0 + (size_t)k * 512);
                acc = __builtin_amdgcn_wmma_f32_16x16x32_bf16(false, a, false, b, (short)0, acc, false, false);
            }
#pragma unroll
            for (int i = 0; i < 8; ++i) {
                float fc = swishf(acc[i]);
                float sp = swishf(m3[i]);              // s3_prev recomputed
                float m  = m3[i] * 0.2f * (1.0f - sp) + fc;
                m3[i] = m;
                sum3[i] += swishf(m);
            }
        }

        wait_async0();                // close out prefetch before next read
        __syncthreads();
    }

    if (wave == 0) {
#pragma unroll
        for (int i = 0; i < 8; ++i) {
            int rowc = i + rowhi;
            if (colw < OUT_DIM)
                out[(size_t)(b0 + rowc) * OUT_DIM + colw] = sum3[i] * (1.0f / 50.0f);
        }
    }
}

extern "C" void kernel_launch(void* const* d_in, const int* in_sizes, int n_in,
                              void* d_out, int out_size, void* d_ws, size_t ws_size,
                              hipStream_t stream) {
    const float* x  = (const float*)d_in[0];
    const float* W1 = (const float*)d_in[1];
    const float* W2 = (const float*)d_in[2];
    const float* W3 = (const float*)d_in[3];
    float* out = (float*)d_out;

    __bf16* wf1 = (__bf16*)d_ws;
    __bf16* wf2 = wf1 + (size_t)NT1 * NK1 * 512;   // 260 frags
    __bf16* wf3 = wf2 + (size_t)NT2 * NK2 * 512;   // 49 frags

    const int f1 = NT1 * NK1, f2 = NT2 * NK2, f3 = NK3;
    snn_pack_weights<<<(f1 * 512 + 255) / 256, 256, 0, stream>>>(W1, wf1, H1, IN_DIM, NK1, f1);
    snn_pack_weights<<<(f2 * 512 + 255) / 256, 256, 0, stream>>>(W2, wf2, H2, H1,     NK2, f2);
    snn_pack_weights<<<(f3 * 512 + 255) / 256, 256, 0, stream>>>(W3, wf3, OUT_DIM, H2, NK3, f3);

    const int B = in_sizes[0] / (T_STEPS * IN_DIM);  // 4096
    snn_forward<<<B / 16, 128, 0, stream>>>(x, wf1, wf2, wf3, out);
}
